// HybridSerialReversedClassifier_16174846836932
// MI455X (gfx1250) — compile-verified, matrix operand order checked by reading, not run on hardware
//
#include <hip/hip_runtime.h>

// ---------------- model dims ----------------
#define B_   128
#define L_   1000
#define C_   12
#define H_   128
#define DM_  128
#define DS_  16
#define DC_  4
#define DI_  256
#define DR_  8
#define FC_  128
#define NC_  5
#define G4H_ 512   // 4*H
#define E2_  512   // 2*DI

// ---------------- WMMA types ----------------
typedef __attribute__((ext_vector_type(16))) __bf16 bf16x16;
typedef __attribute__((ext_vector_type(8)))  float  floatx8;

__device__ __forceinline__ unsigned short f2bf(float f) {
  union { float f; unsigned u; } v; v.f = f;
  unsigned r = v.u + 0x7FFFu + ((v.u >> 16) & 1u);
  return (unsigned short)(r >> 16);
}
__device__ __forceinline__ float bf2f(unsigned short h) {
  union { unsigned u; float f; } v; v.u = ((unsigned)h) << 16; return v.f;
}
__device__ __forceinline__ float sigf(float x) { return 1.f / (1.f + __expf(-x)); }

__device__ __forceinline__ floatx8 zero8() {
  floatx8 z;
#pragma unroll
  for (int i = 0; i < 8; ++i) z[i] = 0.f;
  return z;
}

// A fragment (16x32 bf16, row M = lane&15): lanes 0-15 hold K {k0..k0+7, k0+16..k0+23},
// lanes 16-31 hold K {k0+8..k0+15, k0+24..k0+31}. B fragment from row-major W[N,K] is the
// same pattern with row = n0 + (lane&15). Two 16B loads per lane.
__device__ __forceinline__ bf16x16 frag_ld(const unsigned short* base, int stride,
                                           int row_base, int k0, int lane) {
  const unsigned short* p = base + (size_t)(row_base + (lane & 15)) * (size_t)stride
                                 + k0 + ((lane >> 4) << 3);
  union { bf16x16 v; uint4 q[2]; } u;
  u.q[0] = *reinterpret_cast<const uint4*>(p);
  u.q[1] = *reinterpret_cast<const uint4*>(p + 16);
  return u.v;
}

__device__ __forceinline__ floatx8 bwmma(bf16x16 a, bf16x16 b, floatx8 c) {
  return __builtin_amdgcn_wmma_f32_16x16x32_bf16(false, a, false, b, (short)0, c, false, false);
}

// ---------------- prep kernels (weights -> bf16, fusions) ----------------
__global__ void k_cvt_whh(const float* W_hh, unsigned short* out) {
  int i = blockIdx.x * blockDim.x + threadIdx.x;
  if (i < G4H_ * H_) out[i] = f2bf(W_hh[i]);
}
__global__ void k_build_wih(const float* W_ih, unsigned short* out) {  // pad K 12 -> 32
  int i = blockIdx.x * blockDim.x + threadIdx.x;
  if (i < G4H_ * 32) {
    int r = i >> 5, c = i & 31;
    out[i] = f2bf(c < C_ ? W_ih[r * C_ + c] : 0.f);
  }
}
__global__ void k_bias(const float* b_ih, const float* b_hh, float* out) {
  int i = blockIdx.x * blockDim.x + threadIdx.x;
  if (i < G4H_) out[i] = b_ih[i] + b_hh[i];
}
__global__ void k_build_xproj(const float* xp, unsigned short* out) {  // pad N 40 -> 48
  int i = blockIdx.x * blockDim.x + threadIdx.x;
  if (i < 48 * DI_) {
    int r = i / DI_, c = i % DI_;
    out[i] = f2bf(r < 40 ? xp[r * DI_ + c] : 0.f);
  }
}
// Weff[e,h] = sum_d in_proj_w[e,d] * Wp[d,h]   (fuse input_proj into in_proj, exact)
__global__ void k_weff(const float* in_proj, const float* Wp, unsigned short* out) {
  int i = blockIdx.x * blockDim.x + threadIdx.x;
  if (i >= E2_ * H_) return;
  int e = i >> 7, h = i & 127;
  float s = 0.f;
  for (int d = 0; d < DM_; ++d) s += in_proj[e * DM_ + d] * Wp[d * H_ + h];
  out[i] = f2bf(s);
}
__global__ void k_beff(const float* in_proj, const float* bp, float* out) {
  int e = blockIdx.x * blockDim.x + threadIdx.x;
  if (e >= E2_) return;
  float s = 0.f;
  for (int d = 0; d < DM_; ++d) s += in_proj[e * DM_ + d] * bp[d];
  out[e] = s;
}

// ---------------- persistent LSTM recurrence ----------------
// 8 blocks (batch tile = 16), 256 threads = 8 waves. W_hh/W_ih/bias + h/c/gates in LDS.
// Per step per wave: 4 N-tiles x (1 x-Ktile + 4 h-Ktiles) = 20 WMMAs.
#define SO_WHH  0
#define SO_WIH  131072
#define SO_BIAS 163840
#define SO_H    165888
#define SO_XT   169984
#define SO_C    171008
#define SO_G    179200
#define LSTM_SMEM 211968

__global__ void k_lstm(const float* __restrict__ x,
                       const unsigned short* __restrict__ whh_bf,
                       const unsigned short* __restrict__ wih_bf,
                       const float* __restrict__ bias,
                       unsigned short* __restrict__ hs_bf) {
  extern __shared__ __align__(16) unsigned char smem[];
  unsigned short* sWhh  = (unsigned short*)(smem + SO_WHH);   // 512 x 128 bf16
  unsigned short* sWih  = (unsigned short*)(smem + SO_WIH);   // 512 x 32  bf16
  float*          sBias = (float*)        (smem + SO_BIAS);   // 512 f32
  unsigned short* sH    = (unsigned short*)(smem + SO_H);     // 16 x 128 bf16
  unsigned short* sXt   = (unsigned short*)(smem + SO_XT);    // 16 x 32  bf16
  float*          sC    = (float*)        (smem + SO_C);      // 16 x 128 f32
  float*          sG    = (float*)        (smem + SO_G);      // 16 x 512 f32

  const int tid = threadIdx.x, lane = tid & 31, wave = tid >> 5;
  const int b0 = blockIdx.x * 16;

  for (int i = tid; i < G4H_ * H_; i += 256) sWhh[i] = whh_bf[i];
  for (int i = tid; i < G4H_ * 32; i += 256) sWih[i] = wih_bf[i];
  for (int i = tid; i < G4H_; i += 256)      sBias[i] = bias[i];
  for (int i = tid; i < 16 * H_; i += 256) { sH[i] = 0; sC[i] = 0.f; }
  __syncthreads();

  for (int t = 0; t < L_; ++t) {
    // stage x_t (16 x 12, zero-padded to 32) as bf16
    for (int i = tid; i < 16 * 32; i += 256) {
      int m = i >> 5, c = i & 31;
      float v = (c < C_) ? x[((size_t)(b0 + m) * L_ + t) * C_ + c] : 0.f;
      sXt[i] = f2bf(v);
    }
    __syncthreads();

    // gates(16 x 512) = x_t @ Wih^T + h @ Whh^T ; wave w owns cols [w*64, w*64+64)
    floatx8 acc[4];
#pragma unroll
    for (int nt = 0; nt < 4; ++nt) acc[nt] = zero8();

    bf16x16 ax = frag_ld(sXt, 32, 0, 0, lane);
#pragma unroll
    for (int nt = 0; nt < 4; ++nt) {
      bf16x16 bx = frag_ld(sWih, 32, wave * 64 + nt * 16, 0, lane);
      acc[nt] = bwmma(ax, bx, acc[nt]);
    }
#pragma unroll
    for (int kk = 0; kk < 4; ++kk) {
      bf16x16 ah = frag_ld(sH, H_, 0, kk * 32, lane);
#pragma unroll
      for (int nt = 0; nt < 4; ++nt) {
        bf16x16 bh = frag_ld(sWhh, H_, wave * 64 + nt * 16, kk * 32, lane);
        acc[nt] = bwmma(ah, bh, acc[nt]);
      }
    }
#pragma unroll
    for (int nt = 0; nt < 4; ++nt) {
      int n = wave * 64 + nt * 16 + (lane & 15);
#pragma unroll
      for (int v = 0; v < 8; ++v) {
        int m = v + ((lane >> 4) << 3);
        sG[m * G4H_ + n] = acc[nt][v] + sBias[n];
      }
    }
    __syncthreads();

    // elementwise gate math (PyTorch order i,f,g,o), update c/h, emit h as bf16
    for (int i = tid; i < 16 * H_; i += 256) {
      int m = i >> 7, n = i & 127;
      float gi = sG[m * G4H_ + n];
      float gf = sG[m * G4H_ + H_ + n];
      float gg = sG[m * G4H_ + 2 * H_ + n];
      float go = sG[m * G4H_ + 3 * H_ + n];
      float c  = sigf(gf) * sC[i] + sigf(gi) * tanhf(gg);
      float h  = sigf(go) * tanhf(c);
      sC[i] = c;
      unsigned short hb = f2bf(h);
      sH[i] = hb;
      hs_bf[((size_t)(b0 + m) * L_ + t) * H_ + n] = hb;
    }
    __syncthreads();
  }
}

// ---------------- generic bf16 WMMA GEMM: Out[M,N] = A[M,K] @ W[N,K]^T (+bias) --------
template <int OUT_BF16, int K, int N>
__global__ void k_gemm(const unsigned short* __restrict__ A,
                       const unsigned short* __restrict__ W,
                       const float* __restrict__ bias, void* outp, int M) {
  const int lane = threadIdx.x & 31, wave = threadIdx.x >> 5;
  const int mt = blockIdx.x * 8 + wave;
  if (mt * 16 >= M) return;

  bf16x16 afr[K / 32];
#pragma unroll
  for (int kk = 0; kk < K / 32; ++kk) afr[kk] = frag_ld(A, K, mt * 16, kk * 32, lane);

#pragma unroll 1
  for (int nt = 0; nt < N / 16; ++nt) {
    floatx8 acc = zero8();
#pragma unroll
    for (int kk = 0; kk < K / 32; ++kk) {
      bf16x16 bfr = frag_ld(W, K, nt * 16, kk * 32, lane);
      acc = bwmma(afr[kk], bfr, acc);
    }
    const int col = nt * 16 + (lane & 15);
    const float bv = bias ? bias[col] : 0.f;
#pragma unroll
    for (int v = 0; v < 8; ++v) {
      int row = mt * 16 + v + ((lane >> 4) << 3);
      float val = acc[v] + bv;
      if (OUT_BF16) ((unsigned short*)outp)[(size_t)row * N + col] = f2bf(val);
      else          ((float*)outp)[(size_t)row * N + col] = val;
    }
  }
}

// ---------------- causal depthwise conv (DC=4) + SiLU ----------------
__global__ void k_conv(const unsigned short* __restrict__ xz, const float* __restrict__ cw,
                       const float* __restrict__ cb, unsigned short* __restrict__ xs) {
  size_t idx = (size_t)blockIdx.x * blockDim.x + threadIdx.x;
  if (idx >= (size_t)B_ * L_ * DI_) return;
  int d = (int)(idx % DI_);
  size_t rest = idx / DI_;
  int l = (int)(rest % L_);
  int b = (int)(rest / L_);
  float acc = cb[d];
#pragma unroll
  for (int j = 0; j < DC_; ++j) {
    int lt = l - (DC_ - 1) + j;
    if (lt >= 0) acc += cw[d * DC_ + j] * bf2f(xz[((size_t)b * L_ + lt) * E2_ + d]);
  }
  xs[idx] = f2bf(acc * sigf(acc));  // silu
}

// ---------------- selective scan + gating + time-mean (fused) ----------------
// 128 blocks (one batch each) x 256 threads (one channel d each). State h[16] in VGPRs.
// ybar[b,d] = mean_t ( (h.C + D*x) * silu(z) )  -> out_proj applied afterwards (linear).
__global__ void k_scan(const float* __restrict__ xdbl, const unsigned short* __restrict__ xs,
                       const unsigned short* __restrict__ xz,
                       const float* __restrict__ dtw_g, const float* __restrict__ dtb_g,
                       const float* __restrict__ A_log, const float* __restrict__ Dp,
                       float* __restrict__ ybar) {
  const int d = threadIdx.x, b = blockIdx.x;
  float Arow[DS_];
#pragma unroll
  for (int s = 0; s < DS_; ++s) Arow[s] = -__expf(A_log[d * DS_ + s]);
  float dtw[DR_];
#pragma unroll
  for (int r = 0; r < DR_; ++r) dtw[r] = dtw_g[d * DR_ + r];
  const float dtb = dtb_g[d], Dd = Dp[d];

  float h[DS_];
#pragma unroll
  for (int s = 0; s < DS_; ++s) h[s] = 0.f;
  float ysum = 0.f;

  __shared__ float sBC[48];  // [0..7]=dt  [8..23]=B_t  [24..39]=C_t
  for (int t = 0; t < L_; ++t) {
    if (threadIdx.x < 48) sBC[threadIdx.x] = xdbl[((size_t)b * L_ + t) * 48 + threadIdx.x];
    __syncthreads();

    float dv = dtb;
#pragma unroll
    for (int r = 0; r < DR_; ++r) dv += dtw[r] * sBC[r];
    float delta = (dv > 20.f) ? dv : log1pf(__expf(dv));  // softplus

    float xt = bf2f(xs[((size_t)b * L_ + t) * DI_ + d]);
    float zt = bf2f(xz[((size_t)b * L_ + t) * E2_ + DI_ + d]);
    float dx = delta * xt;
    float y = 0.f;
#pragma unroll
    for (int s = 0; s < DS_; ++s) {
      float dA = __expf(delta * Arow[s]);
      h[s] = dA * h[s] + dx * sBC[8 + s];
      y += h[s] * sBC[24 + s];
    }
    ysum += (y + Dd * xt) * (zt * sigf(zt));
    __syncthreads();
  }
  ybar[b * DI_ + d] = ysum * (1.0f / (float)L_);
}

// ---------------- head: out_proj(pooled) -> fc1+relu -> fc2 ----------------
__global__ void k_pool(const float* __restrict__ ybar, const float* __restrict__ opw,
                       float* __restrict__ pooled) {
  int i = blockIdx.x * blockDim.x + threadIdx.x;
  if (i >= B_ * DM_) return;
  int b = i >> 7, e = i & 127;
  float s = 0.f;
  for (int d = 0; d < DI_; ++d) s += ybar[b * DI_ + d] * opw[e * DI_ + d];
  pooled[i] = s;
}
__global__ void k_fc1(const float* __restrict__ pooled, const float* __restrict__ w,
                      const float* __restrict__ bias, float* __restrict__ hfc) {
  int i = blockIdx.x * blockDim.x + threadIdx.x;
  if (i >= B_ * FC_) return;
  int b = i >> 7, f = i & 127;
  float s = bias[f];
  for (int d = 0; d < DM_; ++d) s += pooled[b * DM_ + d] * w[f * DM_ + d];
  hfc[i] = s > 0.f ? s : 0.f;
}
__global__ void k_fc2(const float* __restrict__ hfc, const float* __restrict__ w,
                      const float* __restrict__ bias, float* __restrict__ out) {
  int i = blockIdx.x * blockDim.x + threadIdx.x;
  if (i >= B_ * NC_) return;
  int b = i / NC_, n = i % NC_;
  float s = bias[n];
  for (int d = 0; d < FC_; ++d) s += hfc[b * FC_ + d] * w[n * FC_ + d];
  out[i] = s;
}

// ---------------- launch ----------------
extern "C" void kernel_launch(void* const* d_in, const int* in_sizes, int n_in,
                              void* d_out, int out_size, void* d_ws, size_t ws_size,
                              hipStream_t stream) {
  (void)in_sizes; (void)n_in; (void)out_size; (void)ws_size;
  const float* x        = (const float*)d_in[0];
  const float* W_ih     = (const float*)d_in[1];
  const float* W_hh     = (const float*)d_in[2];
  const float* b_ih     = (const float*)d_in[3];
  const float* b_hh     = (const float*)d_in[4];
  const float* Wp       = (const float*)d_in[5];
  const float* bp       = (const float*)d_in[6];
  const float* in_proj  = (const float*)d_in[7];
  const float* conv_w   = (const float*)d_in[8];
  const float* conv_b   = (const float*)d_in[9];
  const float* x_proj   = (const float*)d_in[10];
  const float* dt_w     = (const float*)d_in[11];
  const float* dt_b     = (const float*)d_in[12];
  const float* A_log    = (const float*)d_in[13];
  const float* Dp       = (const float*)d_in[14];
  const float* out_proj = (const float*)d_in[15];
  const float* fc1_w    = (const float*)d_in[16];
  const float* fc1_b    = (const float*)d_in[17];
  const float* fc2_w    = (const float*)d_in[18];
  const float* fc2_b    = (const float*)d_in[19];
  float* out = (float*)d_out;

  unsigned char* ws = (unsigned char*)d_ws;
  unsigned short* whh_bf = (unsigned short*)(ws + 0);        // 131072 B
  unsigned short* wih_bf = (unsigned short*)(ws + 131072);   //  32768 B
  float*          bias   = (float*)        (ws + 163840);    //   2048 B
  unsigned short* weff   = (unsigned short*)(ws + 165888);   // 131072 B
  float*          beff   = (float*)        (ws + 296960);    //   2048 B
  unsigned short* xprojp = (unsigned short*)(ws + 299008);   //  24576 B (48x256)
  float*          ybar   = (float*)        (ws + 323584);    // 131072 B
  float*          pooled = (float*)        (ws + 454656);    //  65536 B
  float*          hfc    = (float*)        (ws + 520192);    //  65536 B
  unsigned short* hs     = (unsigned short*)(ws + 585728);                         // 32,768,000 B
  unsigned short* xz     = (unsigned short*)(ws + 585728 + 32768000ull);           // 131,072,000 B
  unsigned short* xs     = (unsigned short*)(ws + 585728 + 163840000ull);          // 65,536,000 B
  float*          xdbl   = (float*)        (ws + 585728 + 229376000ull);           // 24,576,000 B

  // weight prep / fusions
  k_cvt_whh   <<<(G4H_*H_ + 255) / 256, 256, 0, stream>>>(W_hh, whh_bf);
  k_build_wih <<<(G4H_*32  + 255) / 256, 256, 0, stream>>>(W_ih, wih_bf);
  k_bias      <<<2, 256, 0, stream>>>(b_ih, b_hh, bias);
  k_build_xproj<<<(48*DI_ + 255) / 256, 256, 0, stream>>>(x_proj, xprojp);
  k_weff      <<<(E2_*H_ + 255) / 256, 256, 0, stream>>>(in_proj, Wp, weff);
  k_beff      <<<2, 256, 0, stream>>>(in_proj, bp, beff);

  // persistent LSTM recurrence (WMMA, weights in LDS)
  k_lstm<<<B_ / 16, 256, LSTM_SMEM, stream>>>(x, whh_bf, wih_bf, bias, hs);

  // fused (input_proj ∘ in_proj): xz[BL,512] = hs[BL,128] @ Weff^T + beff
  k_gemm<1, 128, 512><<<(B_ * L_) / 128, 256, 0, stream>>>(hs, weff, beff, xz, B_ * L_);

  // causal depthwise conv + silu -> xs
  k_conv<<<(B_ * L_ * DI_) / 256, 256, 0, stream>>>(xz, conv_w, conv_b, xs);

  // x_proj: xdbl[BL,48] = xs[BL,256] @ xprojp^T  (N padded 40->48)
  k_gemm<0, 256, 48><<<(B_ * L_) / 128, 256, 0, stream>>>(xs, xprojp, nullptr, xdbl, B_ * L_);

  // selective scan + gating + time-mean
  k_scan<<<B_, DI_, 0, stream>>>(xdbl, xs, xz, dt_w, dt_b, A_log, Dp, ybar);

  // head (mean commuted before out_proj — exact)
  k_pool<<<(B_ * DM_ + 255) / 256, 256, 0, stream>>>(ybar, out_proj, pooled);
  k_fc1 <<<(B_ * FC_ + 255) / 256, 256, 0, stream>>>(pooled, fc1_w, fc1_b, hfc);
  k_fc2 <<<(B_ * NC_ + 255) / 256, 256, 0, stream>>>(hfc, fc2_w, fc2_b, out);
}
